// SVHunterMultiHeadAttention_5497558139332
// MI455X (gfx1250) — matrix operand order repaired and checked by reading, not compiled, not run on hardware
//
#include <hip/hip_runtime.h>
#include <hip/hip_bf16.h>

typedef _Float16 h16;
typedef __attribute__((ext_vector_type(16))) _Float16 h16x16;
typedef __attribute__((ext_vector_type(4)))  _Float16 h16x4;
typedef __attribute__((ext_vector_type(8)))  float    f32x8;
typedef __attribute__((ext_vector_type(4)))  unsigned int uint32x4;
typedef __attribute__((ext_vector_type(8)))  int      int32x8;
typedef __attribute__((ext_vector_type(4)))  int      int32x4;

union FragAB { h16x16 v; unsigned int u[8]; };

constexpr int kEMB   = 1024;
constexpr int kH     = 16;
constexpr int kDK    = 64;
constexpr int kINNER = 1024;
constexpr int kSEQ   = 2048;
constexpr int kBATCH = 4;
constexpr int kM     = kBATCH * kSEQ;   // 8192 token rows

// ---------------------------------------------------------------------------
// Fragment loaders (CDNA5 ISA 7.12.2, wave32). Each lowers to 2x ds_load_b128.
// ---------------------------------------------------------------------------
__device__ __forceinline__ h16x16 loadA(const h16* p, int ld) {
    const int lane = threadIdx.x & 31;
    const int m = lane & 15, hf = lane >> 4;
    FragAB f;
#pragma unroll
    for (int i = 0; i < 8; ++i) {
        int k = 2 * i + (i >= 4 ? 8 : 0) + 8 * hf;
        f.u[i] = *(const unsigned int*)(p + m * ld + k);
    }
    return f.v;
}

__device__ __forceinline__ h16x16 loadB(const h16* p, int ld) {
    const int lane = threadIdx.x & 31;
    const int n = lane & 15, hf = lane >> 4;
    FragAB f;
#pragma unroll
    for (int i = 0; i < 8; ++i) {
        int k = 2 * i + (i >= 4 ? 8 : 0) + 8 * hf;
        f.u[i] = *(const unsigned int*)(p + n * ld + k);
    }
    return f.v;
}

__device__ __forceinline__ f32x8 wmma_f16(h16x16 a, h16x16 b, f32x8 c) {
    return __builtin_amdgcn_wmma_f32_16x16x32_f16(false, a, false, b,
                                                  (short)0, c, false, false);
}

// ---------------------------------------------------------------------------
// Tensor Data Mover: DMA a 2D fp16 tile (tile0 x tile1 elements, row stride
// stride0 elements) from global memory into LDS at byte offset lds_off.
// Descriptor layout per CDNA5 ISA sections 8.3-8.6 (6-arg builtin form).
// ---------------------------------------------------------------------------
__device__ __forceinline__ void tdm_load_2d(unsigned int lds_off, const void* gptr,
                                            unsigned int tile0, unsigned int tile1,
                                            unsigned long long stride0) {
    unsigned long long ga = (unsigned long long)gptr;
    uint32x4 g0;
    g0.x = 1u;                                                    // count = 1
    g0.y = lds_off;                                               // lds_addr
    g0.z = (unsigned int)(ga & 0xFFFFFFFFull);                    // gaddr[31:0]
    g0.w = (unsigned int)((ga >> 32) & 0x1FFFFFFull) | (2u << 30);// gaddr[56:32]|type=2
    int32x8 g1;
    g1[0] = (int)(1u << 16);                                      // data_size = 2B
    g1[1] = (int)((tile0 & 0xFFFFu) << 16);                       // tensor_dim0 = tile0
    g1[2] = (int)(((tile0 >> 16) & 0xFFFFu) | ((tile1 & 0xFFFFu) << 16)); // tensor_dim1
    g1[3] = (int)(((tile1 >> 16) & 0xFFFFu) | ((tile0 & 0xFFFFu) << 16)); // tile_dim0
    g1[4] = (int)(tile1 & 0xFFFFu);                               // tile_dim1
    g1[5] = (int)(stride0 & 0xFFFFFFFFull);                       // dim0_stride lo
    g1[6] = (int)((stride0 >> 32) & 0xFFFFull);                   // dim0_stride hi
    g1[7] = 0;
    int32x4 z4 = {0, 0, 0, 0};
    int32x8 z8 = {0, 0, 0, 0, 0, 0, 0, 0};
    __builtin_amdgcn_tensor_load_to_lds(g0, g1, z4, z4, z8, 0);
}

__device__ __forceinline__ unsigned int lds_offset(const void* p) {
    return (unsigned int)(unsigned long long)p;   // LDS addr = addr[31:0]
}

// ---------------------------------------------------------------------------
// fp32 -> fp16 conversion (vectorized x4)
// ---------------------------------------------------------------------------
__global__ __launch_bounds__(256) void cvt_f32_f16(const float* __restrict__ src,
                                                   h16* __restrict__ dst, int n4) {
    int i = blockIdx.x * 256 + threadIdx.x;
    if (i < n4) {
        float4 f = ((const float4*)src)[i];
        h16x4 o;
        o.x = (h16)f.x; o.y = (h16)f.y; o.z = (h16)f.z; o.w = (h16)f.w;
        ((h16x4*)dst)[i] = o;
    }
}

// ---------------------------------------------------------------------------
// Shared GEMM stage: acc[8] += A(16x64) * B(64x128), with B-fragment loads
// software-pipelined one j-tile ahead so ds latency overlaps the WMMAs.
// ---------------------------------------------------------------------------
__device__ __forceinline__ void gemm_stage(const h16 (*As)[64], const h16 (*Bs)[64],
                                           int wave, f32x8 acc[8]) {
    const h16* a_base = &As[wave * 16][0];
    h16x16 a0 = loadA(a_base, 64);        // k = 0..31
    h16x16 a1 = loadA(a_base + 32, 64);   // k = 32..63
    h16x16 b0 = loadB(&Bs[0][0], 64);
    h16x16 b1 = loadB(&Bs[0][32], 64);
#pragma unroll
    for (int j = 0; j < 8; ++j) {
        h16x16 nb0, nb1;
        if (j < 7) {
            nb0 = loadB(&Bs[(j + 1) * 16][0], 64);
            nb1 = loadB(&Bs[(j + 1) * 16][32], 64);
        }
        acc[j] = wmma_f16(a0, b0, acc[j]);
        acc[j] = wmma_f16(a1, b1, acc[j]);
        b0 = nb0; b1 = nb1;
    }
}

// ---------------------------------------------------------------------------
// Y[m,d] = sum_e X[m,e]*W[d,e] + bias[d].  Block tile 128(M) x 128(N),
// K staged 64 at a time via TDM double-buffering. 8 waves, wave = 16 rows.
// Output fp16 in [B,H,S,DK] layout.
// ---------------------------------------------------------------------------
__global__ __launch_bounds__(256) void qkv_gemm(const h16* __restrict__ xb,
                                                const h16* __restrict__ wb,
                                                const float* __restrict__ bias,
                                                h16* __restrict__ out) {
    __shared__ h16 As[2][128][64];
    __shared__ h16 Bs[2][128][64];
    const int tid = threadIdx.x, wave = tid >> 5, lane = tid & 31;
    const int hf = lane >> 4, nloc = lane & 15;
    const int m0 = blockIdx.y * 128;
    const int n0 = blockIdx.x * 128;

    f32x8 acc[8];
#pragma unroll
    for (int j = 0; j < 8; ++j)
#pragma unroll
        for (int r = 0; r < 8; ++r) acc[j][r] = 0.0f;

    if (wave == 0) {
        tdm_load_2d(lds_offset(&As[0][0][0]), xb + (size_t)m0 * kEMB, 64, 128, kEMB);
        tdm_load_2d(lds_offset(&Bs[0][0][0]), wb + (size_t)n0 * kEMB, 64, 128, kEMB);
    }

    constexpr int kStages = kEMB / 64;   // 16
    for (int s = 0; s < kStages; ++s) {
        const int buf = s & 1;
        if (s + 1 < kStages) {
            if (wave == 0) {
                const int e1 = (s + 1) * 64;
                tdm_load_2d(lds_offset(&As[buf ^ 1][0][0]),
                            xb + (size_t)m0 * kEMB + e1, 64, 128, kEMB);
                tdm_load_2d(lds_offset(&Bs[buf ^ 1][0][0]),
                            wb + (size_t)n0 * kEMB + e1, 64, 128, kEMB);
            }
            __builtin_amdgcn_s_wait_tensorcnt(2);   // stage s pair complete
        } else {
            __builtin_amdgcn_s_wait_tensorcnt(0);
        }
        __syncthreads();
        gemm_stage(As[buf], Bs[buf], wave, acc);
        __syncthreads();
    }

#pragma unroll
    for (int j = 0; j < 8; ++j) {
        int d = n0 + j * 16 + nloc;
        float bv = bias[d];
        int h = d >> 6, dk = d & 63;
#pragma unroll
        for (int r = 0; r < 8; ++r) {
            int M = m0 + wave * 16 + r + 8 * hf;
            int b = M >> 11, s = M & (kSEQ - 1);
            out[(((size_t)(b * kH + h)) * kSEQ + s) * kDK + dk] = (h16)(acc[j][r] + bv);
        }
    }
}

// ---------------------------------------------------------------------------
// Flash attention: one block = 128 query rows of one (b,h); wave = 16 rows.
// K tile DMA'd by TDM, V tile staged transposed manually (overlaps with TDM).
// ---------------------------------------------------------------------------
__global__ __launch_bounds__(256) void attn(const h16* __restrict__ Qb,
                                            const h16* __restrict__ Kb,
                                            const h16* __restrict__ Vb,
                                            h16* __restrict__ Ob) {
    __shared__ h16 Kt[32][64];       // [kv][d]   (TDM destination)
    __shared__ h16 VtT[64][32];      // [d][kv]   (transposed for B-frag loads)
    __shared__ h16 Pw[8][16][32];    // per-wave P scratch [qrow][kv]

    const int tid = threadIdx.x, wave = tid >> 5, lane = tid & 31;
    const int hf = lane >> 4, nloc = lane & 15;
    const int bh = blockIdx.y;
    const int q0 = blockIdx.x * 128 + wave * 16;
    const size_t base = (size_t)bh * kSEQ * kDK;

    // Q fragments (A-layout), loaded once from global
    h16x16 qf[2];
    {
        const h16* qp = Qb + base + (size_t)q0 * kDK;
#pragma unroll
        for (int kk = 0; kk < 2; ++kk) {
            FragAB f;
#pragma unroll
            for (int i = 0; i < 8; ++i) {
                int k = 2 * i + (i >= 4 ? 8 : 0) + 8 * hf;
                f.u[i] = *(const unsigned int*)(qp + nloc * kDK + kk * 32 + k);
            }
            qf[kk] = f.v;
        }
    }

    f32x8 o[4];
    float mx[8], l[8];
#pragma unroll
    for (int j = 0; j < 4; ++j)
#pragma unroll
        for (int r = 0; r < 8; ++r) o[j][r] = 0.0f;
#pragma unroll
    for (int r = 0; r < 8; ++r) { mx[r] = -1e30f; l[r] = 0.0f; }

    for (int kv0 = 0; kv0 < kSEQ; kv0 += 32) {
        // K tile via TDM (32 rows x 64 d, contiguous rows)
        if (wave == 0)
            tdm_load_2d(lds_offset(&Kt[0][0]),
                        Kb + base + (size_t)kv0 * kDK, 64, 32, kDK);
        // V tile staged transposed while the TDM streams
#pragma unroll
        for (int j = 0; j < 8; ++j) {
            int e = tid + 256 * j;                    // 0..2047
            int kv = e >> 6, d = e & 63;
            VtT[d][kv] = Vb[base + (size_t)(kv0 + kv) * kDK + d];
        }
        __builtin_amdgcn_s_wait_tensorcnt(0);
        __syncthreads();

        // scores: preload all 4 K-frags, then chain the 4 WMMAs
        h16x16 kb[4];
#pragma unroll
        for (int t = 0; t < 2; ++t)
#pragma unroll
            for (int kk = 0; kk < 2; ++kk)
                kb[t * 2 + kk] = loadB(&Kt[t * 16][kk * 32], 64);

        f32x8 sc[2];
#pragma unroll
        for (int t = 0; t < 2; ++t) {
            f32x8 s;
#pragma unroll
            for (int r = 0; r < 8; ++r) s[r] = 0.0f;
            s = wmma_f16(qf[0], kb[t * 2 + 0], s);
            s = wmma_f16(qf[1], kb[t * 2 + 1], s);
#pragma unroll
            for (int r = 0; r < 8; ++r) s[r] *= 0.125f;
            sc[t] = s;
        }

        // online softmax; xor masks < 16 stay within each 16-lane row group
        float mt[8];
#pragma unroll
        for (int r = 0; r < 8; ++r) mt[r] = fmaxf(sc[0][r], sc[1][r]);
#pragma unroll
        for (int d = 1; d < 16; d <<= 1)
#pragma unroll
            for (int r = 0; r < 8; ++r) mt[r] = fmaxf(mt[r], __shfl_xor(mt[r], d, 32));

        float alpha[8], rs[8];
#pragma unroll
        for (int r = 0; r < 8; ++r) {
            float mn = fmaxf(mx[r], mt[r]);
            alpha[r] = __expf(mx[r] - mn);
            mx[r] = mn;
            float p0 = __expf(sc[0][r] - mn);
            float p1 = __expf(sc[1][r] - mn);
            sc[0][r] = p0; sc[1][r] = p1;
            rs[r] = p0 + p1;
        }
#pragma unroll
        for (int d = 1; d < 16; d <<= 1)
#pragma unroll
            for (int r = 0; r < 8; ++r) rs[r] += __shfl_xor(rs[r], d, 32);
#pragma unroll
        for (int r = 0; r < 8; ++r) l[r] = l[r] * alpha[r] + rs[r];
#pragma unroll
        for (int j = 0; j < 4; ++j)
#pragma unroll
            for (int r = 0; r < 8; ++r) o[j][r] *= alpha[r];

        // P (C-layout) -> per-wave LDS scratch -> A-layout
#pragma unroll
        for (int t = 0; t < 2; ++t)
#pragma unroll
            for (int r = 0; r < 8; ++r)
                Pw[wave][r + 8 * hf][t * 16 + nloc] = (h16)sc[t][r];

        h16x16 pf = loadA(&Pw[wave][0][0], 32);
        h16x16 vb4[4];
#pragma unroll
        for (int j = 0; j < 4; ++j)
            vb4[j] = loadB(&VtT[j * 16][0], 32);
#pragma unroll
        for (int j = 0; j < 4; ++j)
            o[j] = wmma_f16(pf, vb4[j], o[j]);
        __syncthreads();
    }

    // normalize and store to [B,S,INNER] fp16
    const int b = bh >> 4, h = bh & 15;
#pragma unroll
    for (int r = 0; r < 8; ++r) {
        float inv = 1.0f / l[r];
        int s = q0 + r + 8 * hf;
        size_t row = ((size_t)b * kSEQ + s) * kINNER + h * kDK;
#pragma unroll
        for (int j = 0; j < 4; ++j)
            Ob[row + j * 16 + nloc] = (h16)(o[j][r] * inv);
    }
}

// ---------------------------------------------------------------------------
// Output projection: Y[m,e] = sum_d Ob[m,d]*Wo[e,d] + bo[e], fp32 out.
// Same 128x128 TDM double-buffered tiling.
// ---------------------------------------------------------------------------
__global__ __launch_bounds__(256) void out_gemm(const h16* __restrict__ ob,
                                                const h16* __restrict__ wob,
                                                const float* __restrict__ bo,
                                                float* __restrict__ y) {
    __shared__ h16 As[2][128][64];
    __shared__ h16 Bs[2][128][64];
    const int tid = threadIdx.x, wave = tid >> 5, lane = tid & 31;
    const int hf = lane >> 4, nloc = lane & 15;
    const int m0 = blockIdx.y * 128;
    const int n0 = blockIdx.x * 128;

    f32x8 acc[8];
#pragma unroll
    for (int j = 0; j < 8; ++j)
#pragma unroll
        for (int r = 0; r < 8; ++r) acc[j][r] = 0.0f;

    if (wave == 0) {
        tdm_load_2d(lds_offset(&As[0][0][0]), ob + (size_t)m0 * kINNER, 64, 128, kINNER);
        tdm_load_2d(lds_offset(&Bs[0][0][0]), wob + (size_t)n0 * kINNER, 64, 128, kINNER);
    }

    constexpr int kStages = kINNER / 64;   // 16
    for (int s = 0; s < kStages; ++s) {
        const int buf = s & 1;
        if (s + 1 < kStages) {
            if (wave == 0) {
                const int e1 = (s + 1) * 64;
                tdm_load_2d(lds_offset(&As[buf ^ 1][0][0]),
                            ob + (size_t)m0 * kINNER + e1, 64, 128, kINNER);
                tdm_load_2d(lds_offset(&Bs[buf ^ 1][0][0]),
                            wob + (size_t)n0 * kINNER + e1, 64, 128, kINNER);
            }
            __builtin_amdgcn_s_wait_tensorcnt(2);
        } else {
            __builtin_amdgcn_s_wait_tensorcnt(0);
        }
        __syncthreads();
        gemm_stage(As[buf], Bs[buf], wave, acc);
        __syncthreads();
    }

#pragma unroll
    for (int j = 0; j < 8; ++j) {
        int e = n0 + j * 16 + nloc;
        float bv = bo[e];
#pragma unroll
        for (int r = 0; r < 8; ++r) {
            int M = m0 + wave * 16 + r + 8 * hf;
            y[(size_t)M * kEMB + e] = acc[j][r] + bv;
        }
    }
}

// ---------------------------------------------------------------------------
extern "C" void kernel_launch(void* const* d_in, const int* in_sizes, int n_in,
                              void* d_out, int out_size, void* d_ws, size_t ws_size,
                              hipStream_t stream) {
    const float* x  = (const float*)d_in[0];
    const float* Wq = (const float*)d_in[1];
    const float* bq = (const float*)d_in[2];
    const float* Wk = (const float*)d_in[3];
    const float* bk = (const float*)d_in[4];
    const float* Wv = (const float*)d_in[5];
    const float* bv = (const float*)d_in[6];
    const float* Wo = (const float*)d_in[7];
    const float* bo = (const float*)d_in[8];
    float* out = (float*)d_out;

    h16* xb  = (h16*)d_ws;
    h16* wqb = xb  + (size_t)kM * kEMB;
    h16* wkb = wqb + (size_t)kINNER * kEMB;
    h16* wvb = wkb + (size_t)kINNER * kEMB;
    h16* wob = wvb + (size_t)kINNER * kEMB;
    h16* Qb  = wob + (size_t)kINNER * kEMB;      // [B,H,S,DK]
    h16* Kb  = Qb  + (size_t)kM * kINNER;
    h16* Vb  = Kb  + (size_t)kM * kINNER;
    h16* Ob  = Vb  + (size_t)kM * kINNER;        // [B,S,INNER]

    int n4x = kM * kEMB / 4;
    int n4w = kINNER * kEMB / 4;
    cvt_f32_f16<<<(n4x + 255) / 256, 256, 0, stream>>>(x,  xb,  n4x);
    cvt_f32_f16<<<(n4w + 255) / 256, 256, 0, stream>>>(Wq, wqb, n4w);
    cvt_f32_f16<<<(n4w + 255) / 256, 256, 0, stream>>>(Wk, wkb, n4w);
    cvt_f32_f16<<<(n4w + 255) / 256, 256, 0, stream>>>(Wv, wvb, n4w);
    cvt_f32_f16<<<(n4w + 255) / 256, 256, 0, stream>>>(Wo, wob, n4w);

    dim3 gemm_grid(kINNER / 128, kM / 128);      // (8, 64)
    qkv_gemm<<<gemm_grid, 256, 0, stream>>>(xb, wqb, bq, Qb);
    qkv_gemm<<<gemm_grid, 256, 0, stream>>>(xb, wkb, bk, Kb);
    qkv_gemm<<<gemm_grid, 256, 0, stream>>>(xb, wvb, bv, Vb);

    dim3 attn_grid(kSEQ / 128, kBATCH * kH);     // (16, 64)
    attn<<<attn_grid, 256, 0, stream>>>(Qb, Kb, Vb, Ob);

    out_gemm<<<gemm_grid, 256, 0, stream>>>(Ob, wob, bo, out);
}
